// Text2SemanticDecoder_29454885716206
// MI455X (gfx1250) — matrix operand order repaired and checked by reading, not compiled, not run on hardware
//
#include <hip/hip_runtime.h>
#include <hip/hip_bf16.h>
#include <stdint.h>

typedef __attribute__((ext_vector_type(16))) __bf16 v16bf;
typedef __attribute__((ext_vector_type(8)))  __bf16 v8bf;
typedef __attribute__((ext_vector_type(8)))  float  v8f;

#define DD   512
#define NHH  16
#define HDD  32
#define NLL  24
#define FFH  2048
#define VV   1025
#define TXX  512
#define TYY  1024
#define SS   1536

__device__ __forceinline__ __bf16 to_bf(float f) { return (__bf16)f; }

__device__ __forceinline__ v16bf comb16(v8bf lo, v8bf hi) {
  return __builtin_shufflevector(lo, hi, 0,1,2,3,4,5,6,7,8,9,10,11,12,13,14,15);
}

__device__ __forceinline__ unsigned lds_off(const void* p) {
  return (unsigned)(uintptr_t)p;   // low 32 bits of generic shared pointer = LDS offset
}

__device__ __forceinline__ void async_cp16(unsigned lds, const void* g) {
  // CDNA5 async copy global -> LDS, 16 bytes per lane, tracked by ASYNCcnt
  asm volatile("global_load_async_to_lds_b128 %0, %1, off"
               :: "v"(lds), "v"((uint64_t)(uintptr_t)g) : "memory");
}

__device__ __forceinline__ void wait_async0() {
  asm volatile("s_wait_asynccnt 0x0" ::: "memory");
}

// xor-butterfly within each 16-lane half on the VALU pipe (v_permlane16_b32),
// keeping the DS pipe free for fragment traffic.
__device__ __forceinline__ float pl16(float v, int s1, int s2) {
  int u = __builtin_bit_cast(int, v);
  u = __builtin_amdgcn_permlane16(u, u, s1, s2, false, false);
  return __builtin_bit_cast(float, u);
}
__device__ __forceinline__ float red_max16(float v) {
  v = fmaxf(v, pl16(v, 0x67452301, 0xEFCDAB89));  // xor 1
  v = fmaxf(v, pl16(v, 0x54761032, 0xDCFE98BA));  // xor 2
  v = fmaxf(v, pl16(v, 0x32107654, 0xBA98FEDC));  // xor 4
  v = fmaxf(v, pl16(v, 0xFEDCBA98, 0x76543210));  // xor 8
  return v;
}
__device__ __forceinline__ float red_sum16(float v) {
  v += pl16(v, 0x67452301, 0xEFCDAB89);
  v += pl16(v, 0x54761032, 0xDCFE98BA);
  v += pl16(v, 0x32107654, 0xBA98FEDC);
  v += pl16(v, 0xFEDCBA98, 0x76543210);
  return v;
}

// -------------------- embedding + positional (fp32 + bf16 copies) --------------------
__global__ void embed_kernel(const float* __restrict__ x, const int* __restrict__ y,
                             const float* __restrict__ emb, const float* __restrict__ alpha,
                             const float* __restrict__ pe, const int* __restrict__ prev,
                             float* __restrict__ src, __bf16* __restrict__ srcbf) {
  int idx = blockIdx.x * 256 + threadIdx.x;
  if (idx >= SS * DD) return;
  int t = idx >> 9;
  int d = idx & 511;
  float v;
  if (t < TXX) {
    v = x[idx];
  } else {
    int ty = t - TXX;
    v = emb[(size_t)y[ty] * DD + d] + pe[(size_t)(prev[0] + ty) * DD + d] * alpha[0];
  }
  src[idx] = v;
  srcbf[idx] = to_bf(v);
}

// -------------------- weight convert: W[N,K] fp32 -> tiled-transposed bf16 --------------------
__global__ __launch_bounds__(256)
void convw_kernel(const float* __restrict__ W, __bf16* __restrict__ out, int N, int K) {
  int ktile = blockIdx.x, ntile = blockIdx.y;
  int KT = K >> 5;
  size_t obase = ((size_t)ntile * KT + ktile) * 2048;
  #pragma unroll
  for (int i = 0; i < 8; ++i) {
    int e = threadIdx.x + i * 256;
    int kk = e >> 6, n64 = e & 63;
    int n = ntile * 64 + n64;
    float v = (n < N) ? W[(size_t)n * K + ktile * 32 + kk] : 0.f;
    out[obase + kk * 64 + n64] = to_bf(v);   // coalesced writes
  }
}

// -------------------- GEMM: C[M,N] = A[M,K](bf16) * Wt(tiled bf16) + bias --------------------
// 256 threads = 8 wave32; WG tile 128x64; wave tile 32x32; 2x-unrolled async double buffering.
// KV mode: fp32 results stream straight into the K/V cache regions of d_out (no fp32 qkv buffer).
template<bool RELU, bool WF, bool WB, bool KV>
__global__ __launch_bounds__(256)
void gemm_bf16_wmma(const __bf16* __restrict__ A, const __bf16* __restrict__ Wt,
                    const float* __restrict__ bias, float* __restrict__ outF,
                    __bf16* __restrict__ outB, float* __restrict__ kout,
                    float* __restrict__ vout, int M, int N, int K) {
  __shared__ __align__(16) __bf16 As[2][128 * 40];  // pitch 40 halfs (80B)
  __shared__ __align__(16) __bf16 Ws[2][32 * 72];   // pitch 72 halfs (144B)
  int tid  = threadIdx.x;
  int lane = tid & 31, wv = tid >> 5;
  int hlf  = lane >> 4, l15 = lane & 15;
  int mBase = blockIdx.y * 128;
  int nt    = blockIdx.x;
  int KT    = K >> 5;
  int mw = (wv & 3) * 32, nw = (wv >> 2) * 32;
  v8f acc[2][2] = {};

  // per-thread async staging chunks (16B each): A = 2 chunks, W = 1 chunk
  int arow0 = tid >> 2,        ac = tid & 3;
  int arow1 = arow0 + 64;
  int wkk   = tid >> 3,        wc = tid & 7;
  const __bf16* Ag0 = A + (size_t)(mBase + arow0) * K + ac * 8;
  const __bf16* Ag1 = A + (size_t)(mBase + arow1) * K + ac * 8;
  const __bf16* Wg  = Wt + (size_t)nt * KT * 2048 + wkk * 64 + wc * 8;
  unsigned la00 = lds_off(&As[0][arow0 * 40 + ac * 8]);
  unsigned la01 = lds_off(&As[0][arow1 * 40 + ac * 8]);
  unsigned lw0  = lds_off(&Ws[0][wkk * 72 + wc * 8]);
  unsigned la10 = lds_off(&As[1][arow0 * 40 + ac * 8]);
  unsigned la11 = lds_off(&As[1][arow1 * 40 + ac * 8]);
  unsigned lw1  = lds_off(&Ws[1][wkk * 72 + wc * 8]);

  auto issue = [&](int kt, unsigned a0, unsigned a1, unsigned w0) {
    async_cp16(a0, Ag0 + (size_t)kt * 32);
    async_cp16(a1, Ag1 + (size_t)kt * 32);
    async_cp16(w0, Wg  + (size_t)kt * 2048);
  };
  auto compute = [&](const __bf16* ab, const __bf16* wb) {
    v16bf af[2], bfr[2];
    #pragma unroll
    for (int tm = 0; tm < 2; ++tm) {
      const __bf16* p = ab + (mw + tm * 16 + l15) * 40 + hlf * 8;
      af[tm] = comb16(*(const v8bf*)p, *(const v8bf*)(p + 16));
    }
    #pragma unroll
    for (int tn = 0; tn < 2; ++tn) {
      #pragma unroll
      for (int s = 0; s < 16; ++s) {
        int k = (s < 8) ? (hlf * 8 + s) : (16 + hlf * 8 + s - 8);
        bfr[tn][s] = wb[k * 72 + nw + tn * 16 + l15];
      }
    }
    #pragma unroll
    for (int tm = 0; tm < 2; ++tm)
      #pragma unroll
      for (int tn = 0; tn < 2; ++tn)
        acc[tm][tn] = __builtin_amdgcn_wmma_f32_16x16x32_bf16(
            false, af[tm], false, bfr[tn], (short)0, acc[tm][tn], false, false);
  };

  issue(0, la00, la01, lw0);   // prologue: tile 0 -> buffer 0
  for (int kt = 0; kt < KT; kt += 2) {      // KT is even (16 or 64)
    wait_async0();
    __syncthreads();
    issue(kt + 1, la10, la11, lw1);         // kt+1 < KT always (kt <= KT-2)
    compute(As[0], Ws[0]);
    wait_async0();
    __syncthreads();
    if (kt + 2 < KT) issue(kt + 2, la00, la01, lw0);
    compute(As[1], Ws[1]);
  }

  #pragma unroll
  for (int tm = 0; tm < 2; ++tm)
    #pragma unroll
    for (int tn = 0; tn < 2; ++tn)
      #pragma unroll
      for (int r = 0; r < 8; ++r) {
        int row = mBase + mw + tm * 16 + hlf * 8 + r;   // always < M (M = 12*128)
        int col = nt * 64 + nw + tn * 16 + l15;
        if (col < N) {
          float v = acc[tm][tn][r];
          if (bias) v += bias[col];
          if (RELU) v = fmaxf(v, 0.f);
          if (WF) outF[(size_t)row * N + col] = v;
          if (WB) outB[(size_t)row * N + col] = to_bf(v);
          if (KV) {                           // region uniform per 64-wide N tile
            int c = col - 512;
            if (c >= 512)      vout[(size_t)row * DD + c - 512] = v;
            else if (c >= 0)   kout[(size_t)row * DD + c]       = v;
          }
        }
      }
}

// -------------------- flash attention (bf16 WMMA, online softmax) --------------------
// grid (S/64, NH); block 128 = 4 wave32; reads bf16 qkv copy, writes bf16 attn.
__global__ __launch_bounds__(128)
void attn_kernel(const __bf16* __restrict__ qkvbf, __bf16* __restrict__ attnbf) {
  __shared__ __align__(16) __bf16 Kt[32 * 40];       // [key][hd]
  __shared__ __align__(16) __bf16 Vs[32 * 40];       // [key][hd]
  __shared__ __align__(16) __bf16 Pb[4 * 16 * 40];   // per-wave P staging
  int tid  = threadIdx.x;
  int lane = tid & 31, wv = tid >> 5;
  int hlf  = lane >> 4, l15 = lane & 15;
  int h     = blockIdx.y;
  int qBase = blockIdx.x * 64;

  int qRow = qBase + wv * 16 + l15;
  const __bf16* qb = qkvbf + (size_t)qRow * (3 * DD) + h * HDD + hlf * 8;
  v16bf qf = comb16(*(const v8bf*)qb, *(const v8bf*)(qb + 16));

  const float scl = 0.17677669529663687f;  // 1/sqrt(32)
  v8f o0 = {}, o1 = {};
  float m_i[8], l_i[8];
  #pragma unroll
  for (int r = 0; r < 8; ++r) { m_i[r] = -1e30f; l_i[r] = 0.f; }

  int kkey = tid & 31, kgrp = tid >> 5;          // K: wide load, b16 stores, no transpose
  int vkey = tid >> 2, vc = tid & 3;             // V: async 16B padded copy
  unsigned vlds = lds_off(&Vs[vkey * 40 + vc * 8]);
  int qMin = qBase + wv * 16;                    // wave-uniform min query row

  int kEnd = (qBase < TXX) ? TXX : (qBase + 64);
  for (int kb = 0; kb < kEnd; kb += 32) {
    async_cp16(vlds, qkvbf + (size_t)(kb + vkey) * (3 * DD) + 1024 + h * HDD + vc * 8);
    v8bf kv = *(const v8bf*)(qkvbf + (size_t)(kb + kkey) * (3 * DD) + 512 + h * HDD + kgrp * 8);
    #pragma unroll
    for (int j = 0; j < 8; ++j) Kt[kkey * 40 + kgrp * 8 + j] = kv[j];
    wait_async0();
    __syncthreads();

    const __bf16* kp0 = &Kt[l15 * 40 + hlf * 8];
    const __bf16* kp1 = &Kt[(16 + l15) * 40 + hlf * 8];
    v16bf bk0 = comb16(*(const v8bf*)kp0, *(const v8bf*)(kp0 + 16));
    v16bf bk1 = comb16(*(const v8bf*)kp1, *(const v8bf*)(kp1 + 16));
    v8f z = {};
    v8f s0 = __builtin_amdgcn_wmma_f32_16x16x32_bf16(false, qf, false, bk0, (short)0, z, false, false);
    v8f s1 = __builtin_amdgcn_wmma_f32_16x16x32_bf16(false, qf, false, bk1, (short)0, z, false, false);

    #pragma unroll
    for (int r = 0; r < 8; ++r) { s0[r] *= scl; s1[r] *= scl; }
    if (!((kb + 31 < TXX) || (kb + 31 <= qMin))) {   // wave-uniform fast path
      #pragma unroll
      for (int r = 0; r < 8; ++r) {
        int q_ = qMin + hlf * 8 + r;
        if (!((kb + l15      < TXX) || (kb + l15      <= q_))) s0[r] = -1e30f;
        if (!((kb + 16 + l15 < TXX) || (kb + 16 + l15 <= q_))) s1[r] = -1e30f;
      }
    }

    float mn[8];
    #pragma unroll
    for (int r = 0; r < 8; ++r) {
      float mx = red_max16(fmaxf(s0[r], s1[r]));       // VALU permlane16 butterfly
      mn[r] = fmaxf(m_i[r], mx);
      float al = __expf(m_i[r] - mn[r]);
      m_i[r] = mn[r];
      l_i[r] *= al;
      o0[r] = o0[r] * al;
      o1[r] = o1[r] * al;
    }
    #pragma unroll
    for (int r = 0; r < 8; ++r) {
      float p0 = __expf(s0[r] - mn[r]);
      float p1 = __expf(s1[r] - mn[r]);
      s0[r] = p0; s1[r] = p1;
      l_i[r] += red_sum16(p0 + p1);
    }

    // C-layout -> A-layout for P via per-wave LDS round trip (same-wave DS ops are in-order)
    __bf16* pw = &Pb[wv * 16 * 40];
    #pragma unroll
    for (int r = 0; r < 8; ++r) {
      pw[(hlf * 8 + r) * 40 + l15]      = to_bf(s0[r]);
      pw[(hlf * 8 + r) * 40 + 16 + l15] = to_bf(s1[r]);
    }
    const __bf16* pp = &pw[l15 * 40 + hlf * 8];
    v16bf pf = comb16(*(const v8bf*)pp, *(const v8bf*)(pp + 16));

    v16bf bv0, bv1;
    #pragma unroll
    for (int s = 0; s < 16; ++s) {
      int k = (s < 8) ? (hlf * 8 + s) : (16 + hlf * 8 + s - 8);
      bv0[s] = Vs[k * 40 + l15];
      bv1[s] = Vs[k * 40 + 16 + l15];
    }
    o0 = __builtin_amdgcn_wmma_f32_16x16x32_bf16(false, pf, false, bv0, (short)0, o0, false, false);
    o1 = __builtin_amdgcn_wmma_f32_16x16x32_bf16(false, pf, false, bv1, (short)0, o1, false, false);
    __syncthreads();
  }

  #pragma unroll
  for (int r = 0; r < 8; ++r) {
    int row = qBase + wv * 16 + hlf * 8 + r;
    float inv = 1.f / l_i[r];
    attnbf[(size_t)row * DD + h * HDD + l15]      = to_bf(o0[r] * inv);
    attnbf[(size_t)row * DD + h * HDD + 16 + l15] = to_bf(o1[r] * inv);
  }
}

// -------------------- residual add + LayerNorm (fp32 + bf16 out) --------------------
__global__ __launch_bounds__(256)
void ln_res_kernel(const float* __restrict__ xin, const float* __restrict__ res,
                   const float* __restrict__ g, const float* __restrict__ b,
                   float* __restrict__ outp, __bf16* __restrict__ outbf) {
  __shared__ float red[16];
  int row = blockIdx.x, tid = threadIdx.x;
  size_t base = (size_t)row * DD;
  float v0 = xin[base + tid]       + res[base + tid];
  float v1 = xin[base + tid + 256] + res[base + tid + 256];
  float s = v0 + v1, q = v0 * v0 + v1 * v1;
  #pragma unroll
  for (int m = 1; m < 32; m <<= 1) { s += __shfl_xor(s, m, 32); q += __shfl_xor(q, m, 32); }
  int wv = tid >> 5, lane = tid & 31;
  if (lane == 0) { red[wv] = s; red[8 + wv] = q; }
  __syncthreads();
  if (tid == 0) {
    float ts = 0.f, tq = 0.f;
    #pragma unroll
    for (int i = 0; i < 8; ++i) { ts += red[i]; tq += red[8 + i]; }
    red[0] = ts; red[1] = tq;
  }
  __syncthreads();
  float mean = red[0] * (1.f / 512.f);
  float var  = red[1] * (1.f / 512.f) - mean * mean;
  float rstd = rsqrtf(var + 1e-5f);
  float o0 = (v0 - mean) * rstd * g[tid]       + b[tid];
  float o1 = (v1 - mean) * rstd * g[tid + 256] + b[tid + 256];
  outp[base + tid] = o0;         outp[base + tid + 256] = o1;
  outbf[base + tid] = to_bf(o0); outbf[base + tid + 256] = to_bf(o1);
}

// -------------------- host launch --------------------
extern "C" void kernel_launch(void* const* d_in, const int* in_sizes, int n_in,
                              void* d_out, int out_size, void* d_ws, size_t ws_size,
                              hipStream_t stream) {
  const float* x     = (const float*)d_in[0];
  const int*   y     = (const int*)d_in[1];
  const int*   prev  = (const int*)d_in[4];
  const float* emb   = (const float*)d_in[6];
  const float* alpha = (const float*)d_in[7];
  const float* pe    = (const float*)d_in[8];
  const float* ipw   = (const float*)d_in[9];
  const float* ipb   = (const float*)d_in[10];
  const float* ow    = (const float*)d_in[11];
  const float* ob    = (const float*)d_in[12];
  const float* g1    = (const float*)d_in[13];
  const float* b1    = (const float*)d_in[14];
  const float* g2    = (const float*)d_in[15];
  const float* b2    = (const float*)d_in[16];
  const float* f1w   = (const float*)d_in[17];
  const float* f1b   = (const float*)d_in[18];
  const float* f2w   = (const float*)d_in[19];
  const float* f2bi  = (const float*)d_in[20];
  const float* prw   = (const float*)d_in[21];

  float* out = (float*)d_out;

  float* wsf = (float*)d_ws;
  float* src = wsf;                        // S*D
  float* tmp = src + (size_t)SS * DD;      // S*D
  __bf16* wsb   = (__bf16*)(tmp + (size_t)SS * DD);
  __bf16* srcbf = wsb;                              // S*D
  __bf16* qkvbf = srcbf + (size_t)SS * DD;          // S*3D
  __bf16* attnbf= qkvbf + (size_t)SS * 3 * DD;      // S*D
  __bf16* ff1bf = attnbf + (size_t)SS * DD;         // S*FF
  __bf16* ipwT  = ff1bf + (size_t)SS * FFH;         // 24*16*2048
  __bf16* owT   = ipwT + (size_t)24 * 16 * 2048;    // 8*16*2048
  __bf16* f1wT  = owT  + (size_t)8 * 16 * 2048;     // 32*16*2048
  __bf16* f2wT  = f1wT + (size_t)32 * 16 * 2048;    // 8*64*2048
  __bf16* prwT  = f2wT + (size_t)8 * 64 * 2048;     // 17*16*2048

  const size_t KOFF = (size_t)SS * VV;
  const size_t VOFF = KOFF + (size_t)NLL * SS * DD;

  embed_kernel<<<dim3((SS * DD) / 256), 256, 0, stream>>>(x, y, emb, alpha, pe, prev, src, srcbf);

  for (int l = 0; l < NLL; ++l) {
    convw_kernel<<<dim3(16, 24), 256, 0, stream>>>(ipw + (size_t)l * 3 * DD * DD, ipwT, 3 * DD, DD);
    convw_kernel<<<dim3(16,  8), 256, 0, stream>>>(ow  + (size_t)l * DD * DD,     owT,  DD, DD);
    convw_kernel<<<dim3(16, 32), 256, 0, stream>>>(f1w + (size_t)l * FFH * DD,    f1wT, FFH, DD);
    convw_kernel<<<dim3(64,  8), 256, 0, stream>>>(f2w + (size_t)l * DD * FFH,    f2wT, DD, FFH);

    // QKV: bf16 for attention + K/V fp32 streamed straight into the cache outputs
    gemm_bf16_wmma<false, false, true, true><<<dim3(24, SS / 128), 256, 0, stream>>>(
        srcbf, ipwT, ipb + (size_t)l * 3 * DD, nullptr, qkvbf,
        out + KOFF + (size_t)l * SS * DD, out + VOFF + (size_t)l * SS * DD, SS, 3 * DD, DD);
    attn_kernel<<<dim3(SS / 64, NHH), 128, 0, stream>>>(qkvbf, attnbf);
    gemm_bf16_wmma<false, true, false, false><<<dim3(8, SS / 128), 256, 0, stream>>>(
        attnbf, owT, ob + (size_t)l * DD, tmp, nullptr, nullptr, nullptr, SS, DD, DD);
    ln_res_kernel<<<dim3(SS), 256, 0, stream>>>(
        src, tmp, g1 + (size_t)l * DD, b1 + (size_t)l * DD, src, srcbf);
    gemm_bf16_wmma<true, false, true, false><<<dim3(32, SS / 128), 256, 0, stream>>>(
        srcbf, f1wT, f1b + (size_t)l * FFH, nullptr, ff1bf, nullptr, nullptr, SS, FFH, DD);
    gemm_bf16_wmma<false, true, false, false><<<dim3(8, SS / 128), 256, 0, stream>>>(
        ff1bf, f2wT, f2bi + (size_t)l * DD, tmp, nullptr, nullptr, nullptr, SS, DD, FFH);
    ln_res_kernel<<<dim3(SS), 256, 0, stream>>>(
        src, tmp, g2 + (size_t)l * DD, b2 + (size_t)l * DD, src, srcbf);
  }

  convw_kernel<<<dim3(16, 17), 256, 0, stream>>>(prw, prwT, VV, DD);
  gemm_bf16_wmma<false, true, false, false><<<dim3(17, SS / 128), 256, 0, stream>>>(
      srcbf, prwT, nullptr, out, nullptr, nullptr, nullptr, SS, VV, DD);
}
